// SequenceEncoder_73632919323278
// MI455X (gfx1250) — matrix operand order repaired and checked by reading, not compiled, not run on hardware
//
#include <hip/hip_runtime.h>
#include <hip/hip_bf16.h>
#include <stdint.h>

// Problem constants (match reference)
#define HDIM   2048
#define G3     6144            // 3*H
#define TTOT   8192
#define TCHUNK 512             // timesteps per igates chunk
#define NCHUNK (TTOT / TCHUNK)

// Scan kernel geometry
#define SCAN_WGS      128
#define IPW           (HDIM / SCAN_WGS)   // 16 hidden indices per WG
#define ROWS_PER_WG   (3 * IPW)           // 48 matvec rows per WG
#define ROWS_PER_WAVE (ROWS_PER_WG / 8)   // 6 rows per wave (8 wave32 / WG)

// Two-level device barrier geometry
#define SYNC_GROUPS   16
#define WGS_PER_GROUP (SCAN_WGS / SYNC_GROUPS)   // 8
// sync word layout (unsigned indices, 256B-separated cachelines):
//   [0]           generation (release word)
//   [64]          root counter
//   [128 + g*64]  group counter g (g = 0..15)
#define SYNC_WORDS    2048                       // 8 KB zeroed region

typedef __bf16 bf16_t;
typedef bf16_t v16bf __attribute__((ext_vector_type(16)));
typedef float  v8f   __attribute__((ext_vector_type(8)));

static __device__ __forceinline__ unsigned short f32_to_bf16_rne(float f) {
  unsigned u = __float_as_uint(f);
  u += 0x7fffu + ((u >> 16) & 1u);   // round-to-nearest-even (finite inputs)
  return (unsigned short)(u >> 16);
}
static __device__ __forceinline__ unsigned pack_bf16x2(float lo, float hi) {
  return (unsigned)f32_to_bf16_rne(lo) | ((unsigned)f32_to_bf16_rne(hi) << 16);
}
static __device__ __forceinline__ float blo(unsigned u) { return __uint_as_float(u << 16); }
static __device__ __forceinline__ float bhi(unsigned u) { return __uint_as_float(u & 0xffff0000u); }

// ---------------------------------------------------------------------------
// Prep: fp32 w_hh -> bf16 copy (stays L2-resident: 25 MB << 192 MB L2)
// ---------------------------------------------------------------------------
__global__ __launch_bounds__(256) void convert_whh_kernel(
    const float* __restrict__ w, unsigned short* __restrict__ o, int n) {
  int i = blockIdx.x * blockDim.x + threadIdx.x;
  int stride = gridDim.x * blockDim.x;
  for (; i < n; i += stride) o[i] = f32_to_bf16_rne(w[i]);
}

__global__ __launch_bounds__(256) void init_state_kernel(
    float* __restrict__ h, unsigned* __restrict__ sync) {
  for (int i = threadIdx.x; i < HDIM; i += blockDim.x) h[i] = 0.0f;
  for (int i = threadIdx.x; i < SYNC_WORDS; i += blockDim.x) sync[i] = 0u;
}

// ---------------------------------------------------------------------------
// igates GEMM for one chunk: ig[m, j] = dot(inputs[t0+m,:], w_ih[j,:]) + bias[j]
// WG = one 16(M) x 128(N) C block: 8 waves share one A tile (staged once by
// the whole WG), each wave owns one 16x16 N-tile. K swept in steps of 32 via
// v_wmma_f32_16x16x32_bf16 (f32 accumulate). Fragments gathered from LDS per
// ISA 7.12.2 VGPR layouts (gathers vectorize to ds_load_b128 pairs).
// ---------------------------------------------------------------------------
__global__ __launch_bounds__(256) void igates_gemm_kernel(
    const float* __restrict__ inp,   // [TTOT, HDIM]
    const float* __restrict__ wih,   // [G3, HDIM]
    const float* __restrict__ bias,  // [G3]
    float* __restrict__ ig,          // [TCHUNK, G3]
    int t0) {
  __shared__ unsigned short aTile[16 * 32];        // shared A tile (bf16)
  __shared__ unsigned short bTile[8][16 * 32];     // per-wave B tile (bf16)

  const int wave = threadIdx.x >> 5;
  const int lane = threadIdx.x & 31;

  const int NGRP = G3 / 128;                 // 48 N-groups
  const int m0   = (blockIdx.x / NGRP) * 16; // chunk-local time row
  const int n0   = (blockIdx.x % NGRP) * 128 + wave * 16;

  const int half = lane >> 4;                // 0: lanes 0-15, 1: lanes 16-31
  const int r    = lane & 15;

  // A staging map: thread tid packs one u32 (two bf16): row tid>>4, K-pair tid&15
  const int arow = threadIdx.x >> 4;
  const int acp  = threadIdx.x & 15;
  // B staging map (per wave): lane loads row lane>>1, 16 floats at col (lane&1)*16
  const int lrow = lane >> 1;
  const int lcol = (lane & 1) * 16;

  unsigned short* bT = bTile[wave];
  v8f c = {0.f, 0.f, 0.f, 0.f, 0.f, 0.f, 0.f, 0.f};

  const float* aBase = inp + (size_t)(t0 + m0) * HDIM;
  const float* bBase = wih + (size_t)n0 * HDIM;

  for (int k0 = 0; k0 < HDIM; k0 += 32) {
    // --- stage A (whole WG, 512 bf16) ---
    {
      const float* ap = aBase + (size_t)arow * HDIM + k0 + 2 * acp;
      float2 av = *(const float2*)ap;
      ((unsigned*)aTile)[arow * 16 + acp] = pack_bf16x2(av.x, av.y);
      if (k0 + 32 < HDIM) __builtin_prefetch(ap + 32, 0, 1);
    }
    // --- stage B (per wave, 16x32 fp32 -> bf16) ---
    {
      const float* bp = bBase + (size_t)lrow * HDIM + k0 + lcol;
#pragma unroll
      for (int j = 0; j < 16; j += 4) {
        float4 bv = *(const float4*)(bp + j);
        int o = (lrow * 32 + lcol + j) >> 1;
        ((unsigned*)bT)[o]     = pack_bf16x2(bv.x, bv.y);
        ((unsigned*)bT)[o + 1] = pack_bf16x2(bv.z, bv.w);
      }
      if (k0 + 32 < HDIM) __builtin_prefetch(bp + 32, 0, 1);
    }
    __syncthreads();   // A visible to all waves (also orders B via dscnt wait)
    __asm__ volatile("s_wait_dscnt 0" ::: "memory");

    // --- gather per-lane fragments (ISA 7.12.2 layouts) ---
    union { v16bf v; unsigned u[8]; } A, B;
    const unsigned* a32 = (const unsigned*)aTile;  // [16][16] u32 view
    const unsigned* b32 = (const unsigned*)bT;
#pragma unroll
    for (int j = 0; j < 8; ++j) {
      // A (16x32 MxK): lane<16 row r holds K {0..7,16..23}; lane>=16 {8..15,24..31}
      int ak = (j & 3) + ((j >> 2) << 3) + (half << 2);   // u32 column = K/2
      A.u[j] = a32[r * 16 + ak];
      // B (32x16 KxN) stored N-major: lane<16 col r holds K {0..7,16..23}; lane>=16 {8..15,24..31}
      B.u[j] = b32[r * 16 + ak];
    }
    c = __builtin_amdgcn_wmma_f32_16x16x32_bf16(false, A.v, false, B.v,
                                                (short)0, c, false, false);
    __syncthreads();   // WAR: protect shared A before next iteration's staging
  }

  const float bn = bias[n0 + r];
#pragma unroll
  for (int v = 0; v < 8; ++v) {
    // C layout: VGPR v, lanes 0-15 -> M=v, lanes 16-31 -> M=v+8; N = lane&15
    int m = v + half * 8;
    ig[(size_t)(m0 + m) * G3 + (n0 + r)] = c[v] + bn;
  }
}

// ---------------------------------------------------------------------------
// Two-level device-wide sense barrier (SCAN_WGS co-resident WGs).
// Arrival: 8 WGs -> group cacheline, 16 groups -> root; release: gen word.
// ---------------------------------------------------------------------------
static __device__ __forceinline__ void grid_barrier(unsigned* sync, int g) {
  __syncthreads();
  if (threadIdx.x == 0) {
    __threadfence();
    unsigned* gen  = sync;
    unsigned* root = sync + 64;
    unsigned* grpc = sync + 128 + (g / WGS_PER_GROUP) * 64;
    const unsigned my =
        __hip_atomic_load(gen, __ATOMIC_RELAXED, __HIP_MEMORY_SCOPE_AGENT);
    bool release = false;
    if (atomicAdd(grpc, 1u) == (unsigned)(WGS_PER_GROUP - 1)) {
      __hip_atomic_store(grpc, 0u, __ATOMIC_RELAXED, __HIP_MEMORY_SCOPE_AGENT);
      if (atomicAdd(root, 1u) == (unsigned)(SYNC_GROUPS - 1)) {
        __hip_atomic_store(root, 0u, __ATOMIC_RELAXED, __HIP_MEMORY_SCOPE_AGENT);
        __threadfence();
        atomicAdd(gen, 1u);    // release
        release = true;
      }
    }
    if (!release) {
      while (__hip_atomic_load(gen, __ATOMIC_RELAXED, __HIP_MEMORY_SCOPE_AGENT) == my) {
        __builtin_amdgcn_s_sleep(1);
      }
      __threadfence();         // acquire
    }
  }
  __syncthreads();
}

// ---------------------------------------------------------------------------
// Persistent GRU scan over one chunk. WG g owns hidden indices
// i = g*16 .. g*16+15 and computes w_hh rows {i, 2048+i, 4096+i} so the
// elementwise gate update is WG-local; one device barrier per timestep.
// w_hh streamed as bf16 from L2 (25 MB/step), h broadcast via global+LDS.
// ---------------------------------------------------------------------------
__global__ __launch_bounds__(256) void gru_scan_kernel(
    const unsigned short* __restrict__ whh,  // [G3, HDIM] bf16
    const float* __restrict__ ig,            // [TCHUNK, G3]
    const int* __restrict__ mask,            // [TTOT]
    const float* __restrict__ bias_n,        // [HDIM]
    float* __restrict__ hbuf,                // [HDIM] carried hidden state
    float* __restrict__ out,                 // [TTOT*HDIM + HDIM]
    unsigned* __restrict__ sync,             // SYNC_WORDS region
    int t0) {
  __shared__ float hprev[HDIM];
  __shared__ float dots[ROWS_PER_WG];

  const int g    = blockIdx.x;
  const int wave = threadIdx.x >> 5;
  const int lane = threadIdx.x & 31;

  int rowoff[ROWS_PER_WAVE];
#pragma unroll
  for (int rr = 0; rr < ROWS_PER_WAVE; ++rr) {
    int lr = wave * ROWS_PER_WAVE + rr;                      // 0..47
    rowoff[rr] = ((lr >> 4) * HDIM + g * IPW + (lr & 15));   // global row in w_hh
  }

  for (int s = 0; s < TCHUNK; ++s) {
    const int t = t0 + s;

    // broadcast current h into LDS
    {
      int base = threadIdx.x * 8;
      float4 h0 = *(const float4*)(hbuf + base);
      float4 h1 = *(const float4*)(hbuf + base + 4);
      *(float4*)(hprev + base)     = h0;
      *(float4*)(hprev + base + 4) = h1;
    }
    __syncthreads();

    // 6 dot products of length 2048 per wave (bf16 weights, fp32 math)
    float acc[ROWS_PER_WAVE];
#pragma unroll
    for (int rr = 0; rr < ROWS_PER_WAVE; ++rr) acc[rr] = 0.0f;

    for (int it = 0; it < 8; ++it) {
      int base = it * 256 + lane * 8;
      float4 hv0 = *(const float4*)(hprev + base);
      float4 hv1 = *(const float4*)(hprev + base + 4);
#pragma unroll
      for (int rr = 0; rr < ROWS_PER_WAVE; ++rr) {
        const uint4 w = *(const uint4*)(whh + (size_t)rowoff[rr] * HDIM + base);
        float a = acc[rr];
        a = fmaf(blo(w.x), hv0.x, a); a = fmaf(bhi(w.x), hv0.y, a);
        a = fmaf(blo(w.y), hv0.z, a); a = fmaf(bhi(w.y), hv0.w, a);
        a = fmaf(blo(w.z), hv1.x, a); a = fmaf(bhi(w.z), hv1.y, a);
        a = fmaf(blo(w.w), hv1.z, a); a = fmaf(bhi(w.w), hv1.w, a);
        acc[rr] = a;
      }
    }
#pragma unroll
    for (int rr = 0; rr < ROWS_PER_WAVE; ++rr) {
      float v = acc[rr];
#pragma unroll
      for (int off = 16; off > 0; off >>= 1) v += __shfl_xor(v, off, 32);  // wave32
      if (lane == 0) dots[wave * ROWS_PER_WAVE + rr] = v;
    }
    __syncthreads();

    // elementwise GRU update for this WG's 16 hidden indices
    if (threadIdx.x < IPW) {
      const int il = threadIdx.x;
      const int i  = g * IPW + il;
      const float hr = dots[il];
      const float hz = dots[IPW + il];
      const float hn = dots[2 * IPW + il];
      const float* igt = ig + (size_t)s * G3;
      const float ir  = igt[i];
      const float iz  = igt[HDIM + i];
      const float inn = igt[2 * HDIM + i];
      const float rg = 1.0f / (1.0f + __expf(-(ir + hr)));
      const float zg = 1.0f / (1.0f + __expf(-(iz + hz)));
      const float ng = tanhf(inn + rg * (hn + bias_n[i]));
      const float hp = hprev[i];
      const float prop = ng + zg * (hp - ng);
      const float hnew = (mask[t] != 0) ? prop : hp;
      hbuf[i] = hnew;
      out[(size_t)t * HDIM + i] = hnew;
      if (t == TTOT - 1) out[(size_t)TTOT * HDIM + i] = hnew;  // final_state
      __threadfence();  // make hbuf/out stores device-visible before arriving
    }
    grid_barrier(sync, g);
  }
}

// ---------------------------------------------------------------------------
extern "C" void kernel_launch(void* const* d_in, const int* in_sizes, int n_in,
                              void* d_out, int out_size, void* d_ws, size_t ws_size,
                              hipStream_t stream) {
  const float* inputs = (const float*)d_in[0];   // [T, H]
  const int*   mask   = (const int*)d_in[1];     // [T]
  const float* w_ih   = (const float*)d_in[2];   // [3H, H]
  const float* w_hh   = (const float*)d_in[3];   // [3H, H]
  const float* bias   = (const float*)d_in[4];   // [3H]
  const float* bias_n = (const float*)d_in[5];   // [H]
  float* out = (float*)d_out;                    // [T*H + H]

  char* ws = (char*)d_ws;
  size_t off = 0;
  unsigned short* whhb = (unsigned short*)(ws + off);
  off += (size_t)G3 * HDIM * sizeof(unsigned short);          // 25.2 MB
  float* hbuf = (float*)(ws + off);
  off += (size_t)HDIM * sizeof(float);
  off = (off + 255) & ~(size_t)255;
  unsigned* syncw = (unsigned*)(ws + off);
  off += (size_t)SYNC_WORDS * sizeof(unsigned);               // 8 KB
  float* igchunk = (float*)(ws + off);
  off += (size_t)TCHUNK * G3 * sizeof(float);                 // 12.6 MB

  (void)in_sizes; (void)n_in; (void)out_size; (void)ws_size;

  convert_whh_kernel<<<3072, 256, 0, stream>>>(w_hh, whhb, G3 * HDIM);
  init_state_kernel<<<1, 256, 0, stream>>>(hbuf, syncw);

  const int gemm_blocks = (TCHUNK / 16) * (G3 / 128);         // 32 * 48 = 1536
  for (int c = 0; c < NCHUNK; ++c) {
    const int t0 = c * TCHUNK;
    igates_gemm_kernel<<<gemm_blocks, 256, 0, stream>>>(inputs, w_ih, bias, igchunk, t0);
    gru_scan_kernel<<<SCAN_WGS, 256, 0, stream>>>(whhb, igchunk, mask, bias_n,
                                                  hbuf, out, syncw, t0);
  }
}
